// MoEVectorField_51762945851986
// MI455X (gfx1250) — matrix operand order repaired and checked by reading, not compiled, not run on hardware
//
#include <hip/hip_runtime.h>
#include <hip/hip_bf16.h>

// ---------------------------------------------------------------------------
// Problem constants
// ---------------------------------------------------------------------------
static constexpr int B = 16384;
static constexpr int D = 256;
static constexpr int H = 1024;
static constexpr int E = 8;
static constexpr int KPAD = 288;          // D+1 padded up to multiple of 32
static constexpr int KT1 = KPAD / 32;     // 9  k-tiles for layer 1 / router
static constexpr int KT2 = H / 32;        // 32 k-tiles for layers 2,3
static constexpr int NT_H = H / 16;       // 64 n-tiles (H output)
static constexpr int NT_D = D / 16;       // 16 n-tiles (D output)
static constexpr int TM   = 32;           // routed tokens per expert block

// ---------------------------------------------------------------------------
// Scheduling / CDNA5 feature detection
// ---------------------------------------------------------------------------
#if defined(__has_builtin)
#if __has_builtin(__builtin_amdgcn_sched_barrier)
#define SCHED_BARRIER() __builtin_amdgcn_sched_barrier(0)
#endif
#if __has_builtin(__builtin_amdgcn_tensor_load_to_lds) && \
    __has_builtin(__builtin_amdgcn_s_wait_tensorcnt)
#define HAVE_TDM 1
#endif
#endif
#ifndef SCHED_BARRIER
#define SCHED_BARRIER()
#endif

// ---------------------------------------------------------------------------
// WMMA fragment types (wave32, 16x16x32 bf16 -> f32)
// ---------------------------------------------------------------------------
typedef __attribute__((ext_vector_type(16))) __bf16        v16bf;
typedef __attribute__((ext_vector_type(8)))  float         v8f;
typedef __attribute__((ext_vector_type(4)))  unsigned int  v4u;

union Frag {
    v4u   q[2];
    v16bf v;
};

__device__ __forceinline__ unsigned short f2bf(float f) {
    unsigned int u = __float_as_uint(f);
    unsigned int r = u + 0x7fffu + ((u >> 16) & 1u);   // round-to-nearest-even
    return (unsigned short)(r >> 16);
}
__device__ __forceinline__ float bf2f(unsigned short s) {
    return __uint_as_float(((unsigned int)s) << 16);
}
__device__ __forceinline__ float silu_f(float x) {
    return x / (1.0f + __expf(-x));
}

__device__ __forceinline__ v8f wmma_bf16(v16bf a, v16bf b, v8f c) {
    return __builtin_amdgcn_wmma_f32_16x16x32_bf16(false, a, false, b,
                                                   (short)0, c, false, false);
}

// A fragment (16x32 bf16, row-major source). Lane L: row = L&15,
// 16B at K = kt*32 + (L>>4)*8 and 16B at K+16   (matches ISA A layout).
__device__ __forceinline__ v16bf load_a_global(const unsigned short* A,
                                               long rowBase, int kt, int lane) {
    int koff = kt * 32 + ((lane >> 4) << 3);
    Frag f;
    f.q[0] = *reinterpret_cast<const v4u*>(A + rowBase + koff);
    f.q[1] = *reinterpret_cast<const v4u*>(A + rowBase + koff + 16);
    return f.v;
}
__device__ __forceinline__ v16bf load_a_lds(const unsigned short* S, int lda,
                                            int row0, int kt, int lane) {
    int row  = row0 + (lane & 15);
    int koff = kt * 32 + ((lane >> 4) << 3);
    Frag f;
    f.q[0] = *reinterpret_cast<const v4u*>(S + row * lda + koff);
    f.q[1] = *reinterpret_cast<const v4u*>(S + row * lda + koff + 16);
    return f.v;
}
// B fragment from the pre-packed weight stream: tileIdx = (e*KT+kt)*NT+nt
__device__ __forceinline__ v16bf load_b(const unsigned int* P,
                                        long tileIdx, int lane) {
    const v4u* p = reinterpret_cast<const v4u*>(P + ((tileIdx << 5) + lane) * 8);
    Frag f;
    f.q[0] = p[0];
    f.q[1] = p[1];
    return f.v;
}

// ---------------------------------------------------------------------------
// TDM gather: load up to 16 scattered rows (KPAD bf16 each) of xt into LDS.
// D# per CDNA5 ISA (gather mode, 16-bit indices).
// ---------------------------------------------------------------------------
#ifdef HAVE_TDM
typedef __attribute__((ext_vector_type(4))) unsigned int u32x4;
typedef __attribute__((ext_vector_type(4))) int          i32x4;
typedef __attribute__((ext_vector_type(8))) int          i32x8;
typedef __attribute__((address_space(3))) const char     lds_cchar;

__device__ __forceinline__ unsigned lds_offset(const void* p) {
    return (unsigned)(unsigned long long)(lds_cchar*)p;
}

__device__ __forceinline__ void tdm_gather_rows(const unsigned short* gbase,
                                                unsigned ldsAddr,
                                                const int* __restrict__ toks,
                                                int idx0, int nvalid) {
    unsigned pk[8];
#pragma unroll
    for (int j = 0; j < 8; ++j) {
        int a = (2 * j     < nvalid) ? toks[idx0 + 2 * j]     : 0;
        int b = (2 * j + 1 < nvalid) ? toks[idx0 + 2 * j + 1] : 0;
        pk[j] = (unsigned)(a & 0xFFFF) | ((unsigned)(b & 0xFFFF) << 16);
    }
    unsigned long long ga = (unsigned long long)(const void*)gbase;
    u32x4 g0;
    g0[0] = 1u | (1u << 31);                       // count=1, gather, 16b idx
    g0[1] = ldsAddr;                               // LDS byte address
    g0[2] = (unsigned)(ga & 0xFFFFFFFFu);          // global_addr[31:0]
    g0[3] = (unsigned)((ga >> 32) & 0x1FFFFFFu) | (2u << 30);   // type=2
    i32x8 g1;
    g1[0] = (1 << 16);                             // data_size = 2 bytes
    g1[1] = (KPAD << 16);                          // tensor_dim0 = 288
    g1[2] = (B << 16) & 0xFFFF0000;                // tensor_dim1 = 16384 (lo16)
    g1[3] = (KPAD << 16);                          // tile_dim0 = 288
    g1[4] = nvalid & 0xFFFF;                       // tile_dim1 = #indices
    g1[5] = KPAD;                                  // tensor_dim0_stride = 288
    g1[6] = 0;
    g1[7] = 0;
    i32x4 g2; g2[0] = (int)pk[0]; g2[1] = (int)pk[1]; g2[2] = (int)pk[2]; g2[3] = (int)pk[3];
    i32x4 g3; g3[0] = (int)pk[4]; g3[1] = (int)pk[5]; g3[2] = (int)pk[6]; g3[3] = (int)pk[7];
#if __clang_major__ >= 23
    i32x8 g4 = {0, 0, 0, 0, 0, 0, 0, 0};
    __builtin_amdgcn_tensor_load_to_lds(g0, g1, g2, g3, g4, 0);
#else
    __builtin_amdgcn_tensor_load_to_lds(g0, g1, g2, g3, 0);
#endif
}
#endif  // HAVE_TDM

// ---------------------------------------------------------------------------
// Prep kernels
// ---------------------------------------------------------------------------
__global__ void k_zero(float* out, int* counts, long n) {
    long tid = (long)blockIdx.x * blockDim.x + threadIdx.x;
    if (tid < n) out[tid] = 0.0f;
    if (tid < E) counts[tid] = 0;
}

__global__ void k_pad_xt(const float* __restrict__ x, const float* __restrict__ t,
                         unsigned short* __restrict__ xt, long total) {
    long tid = (long)blockIdx.x * blockDim.x + threadIdx.x;
    if (tid >= total) return;
    int  k = (int)(tid % KPAD);
    long b = tid / KPAD;
    float f = (k < D) ? x[b * D + k] : ((k == D) ? t[b] : 0.0f);
    xt[tid] = f2bf(f);
}

// Repack fp32 weights (E, Ksrc, N) -> bf16x2 stream in exact B-fragment order.
__global__ void k_repack(const float* __restrict__ src, unsigned int* __restrict__ dst,
                         int Ksrc, int KT, int NT, long total) {
    long tid = (long)blockIdx.x * blockDim.x + threadIdx.x;
    if (tid >= total) return;
    int  v    = (int)(tid & 7);
    int  lane = (int)((tid >> 3) & 31);
    long rest = tid >> 8;
    int  nt   = (int)(rest % NT);  rest /= NT;
    int  kt   = (int)(rest % KT);  rest /= KT;
    int  e    = (int)rest;
    int  N    = NT * 16;
    int  col  = nt * 16 + (lane & 15);
    int  k    = kt * 32 + ((lane >> 4) << 3) + (v << 1) + ((v >> 2) << 3);
    float f0 = (k     < Ksrc) ? src[((long)e * Ksrc + k    ) * N + col] : 0.0f;
    float f1 = (k + 1 < Ksrc) ? src[((long)e * Ksrc + k + 1) * N + col] : 0.0f;
    dst[tid] = (unsigned int)f2bf(f0) | ((unsigned int)f2bf(f1) << 16);
}

// ---------------------------------------------------------------------------
// Router GEMM1: h = silu(xt @ Rw1 + Rb1). Double-buffered k-loop with
// sched_barrier fencing so next-kt loads overlap the current WMMA burst.
// ---------------------------------------------------------------------------
__global__ void __launch_bounds__(256) __attribute__((amdgpu_waves_per_eu(1)))
k_router_gemm(const unsigned short* __restrict__ xt, const unsigned int* __restrict__ W,
              const float* __restrict__ bias, unsigned short* __restrict__ h) {
    int m0   = blockIdx.x * 16;
    int w    = threadIdx.x >> 5;
    int lane = threadIdx.x & 31;
    int col  = lane & 15;
    int rbase = (lane >> 4) * 8;
    long rowBase = (long)(m0 + (lane & 15)) * KPAD;

#pragma unroll 1
    for (int nh = 0; nh < 2; ++nh) {
        int nb = w * 8 + nh * 4;
        v8f acc[4];
#pragma unroll
        for (int i = 0; i < 4; ++i) {
            float bv = bias[(nb + i) * 16 + col];
#pragma unroll
            for (int j = 0; j < 8; ++j) acc[i][j] = bv;
        }
        v16bf ac = load_a_global(xt, rowBase, 0, lane);
        v16bf bc[4];
#pragma unroll
        for (int i = 0; i < 4; ++i) bc[i] = load_b(W, (long)0 * NT_H + nb + i, lane);
#pragma unroll 2
        for (int kt = 0; kt < KT1; ++kt) {
            int ktn = (kt + 1 < KT1) ? kt + 1 : kt;
            v16bf an = load_a_global(xt, rowBase, ktn, lane);
            v16bf bn[4];
#pragma unroll
            for (int i = 0; i < 4; ++i) bn[i] = load_b(W, (long)ktn * NT_H + nb + i, lane);
            SCHED_BARRIER();
#pragma unroll
            for (int i = 0; i < 4; ++i) acc[i] = wmma_bf16(ac, bc[i], acc[i]);
            SCHED_BARRIER();
            ac = an;
#pragma unroll
            for (int i = 0; i < 4; ++i) bc[i] = bn[i];
        }
#pragma unroll
        for (int i = 0; i < 4; ++i) {
            int n = (nb + i) * 16 + col;
#pragma unroll
            for (int v = 0; v < 8; ++v)
                h[(long)(m0 + rbase + v) * H + n] = f2bf(silu_f(acc[i][v]));
        }
    }
}

// ---------------------------------------------------------------------------
// Router GEMM2: logits = h @ Rw2 + Rb2 (N=8 -> VALU reduction per token)
// ---------------------------------------------------------------------------
__global__ void __launch_bounds__(256)
k_logits(const unsigned short* __restrict__ h, const float* __restrict__ Rw2,
         const float* __restrict__ Rb2, float* __restrict__ logits) {
    __shared__ float red[256][E];
    __shared__ float red2[64];
    int b = blockIdx.x, tid = threadIdx.x;
    float p[E];
#pragma unroll
    for (int e = 0; e < E; ++e) p[e] = 0.0f;
    for (int i = tid; i < H; i += 256) {
        float hv = bf2f(h[(long)b * H + i]);
#pragma unroll
        for (int e = 0; e < E; ++e) p[e] += hv * Rw2[i * E + e];
    }
#pragma unroll
    for (int e = 0; e < E; ++e) red[tid][e] = p[e];
    __syncthreads();
    if (tid < 64) {
        int e2 = tid & 7, chunk = tid >> 3;
        float s = 0.0f;
        for (int i = 0; i < 32; ++i) s += red[chunk * 32 + i][e2];
        red2[tid] = s;
    }
    __syncthreads();
    if (tid < E) {
        float s = Rb2[tid];
#pragma unroll
        for (int c = 0; c < 8; ++c) s += red2[c * 8 + tid];
        logits[(long)b * E + tid] = s;
    }
}

// ---------------------------------------------------------------------------
// Top-2 gating + scatter into per-expert token lists
// ---------------------------------------------------------------------------
__global__ void k_topk(const float* __restrict__ logits, int* __restrict__ toks,
                       float* __restrict__ alphas, int* __restrict__ counts) {
    int b = blockIdx.x * blockDim.x + threadIdx.x;
    if (b >= B) return;
    float l[E];
#pragma unroll
    for (int e = 0; e < E; ++e) l[e] = logits[(long)b * E + e];
    int i0 = 0;
#pragma unroll
    for (int e = 1; e < E; ++e) if (l[e] > l[i0]) i0 = e;
    int i1 = (i0 == 0) ? 1 : 0;
#pragma unroll
    for (int e = 0; e < E; ++e) if (e != i0 && l[e] > l[i1]) i1 = e;
    float ex = __expf(l[i1] - l[i0]);
    float s  = 1.0f + ex;
    float a0 = 1.0f / s, a1 = ex / s;
    int p0 = atomicAdd(&counts[i0], 1);
    toks[i0 * B + p0] = b;  alphas[i0 * B + p0] = a0;
    int p1 = atomicAdd(&counts[i1], 1);
    toks[i1 * B + p1] = b;  alphas[i1 * B + p1] = a1;
}

// ---------------------------------------------------------------------------
// Fused routed expert MLP. 32 tokens / block. Token rows staged into LDS via
// TDM gather descriptors (aliases the h2 region, which is dead until GEMM2).
// All three GEMMs use double-buffered k-loops with sched_barrier fencing.
// ---------------------------------------------------------------------------
__global__ void __launch_bounds__(256) __attribute__((amdgpu_waves_per_eu(1)))
k_expert(const unsigned short* __restrict__ xt,
         const unsigned int* __restrict__ W1, const float* __restrict__ b1,
         const unsigned int* __restrict__ W2, const float* __restrict__ b2,
         const unsigned int* __restrict__ W3, const float* __restrict__ b3,
         const int* __restrict__ toks_g, const float* __restrict__ alpha_g,
         const int* __restrict__ counts, float* __restrict__ out) {
    int e    = blockIdx.x >> 9;          // 512 worst-case tiles per expert
    int tile = blockIdx.x & 511;
    int cnt  = counts[e];
    int t0   = tile * TM;
    if (t0 >= cnt) return;               // uniform early exit, before barriers

    extern __shared__ unsigned short smem[];
    unsigned short* h2s  = smem;                  // 32 x 1024 bf16 (64 KB)
    unsigned short* h_xs = smem;                  // staged xt tile, aliases h2s
    unsigned short* h1s  = smem + TM * H;         // 32 x 1024 bf16 (64 KB)
    float*          alph = reinterpret_cast<float*>(smem + 2 * TM * H);
    int*            rows = reinterpret_cast<int*>(alph + TM);

    int tid = threadIdx.x;
    if (tid < TM) {
        int idx = t0 + tid;
        bool ok = idx < cnt;
        rows[tid] = ok ? toks_g[e * B + idx] : 0;
        alph[tid] = ok ? alpha_g[e * B + idx] : 0.0f;
    }
    __syncthreads();

    int w = tid >> 5, lane = tid & 31;
    int col = lane & 15, rbase = (lane >> 4) * 8;

    // ---- stage gathered xt rows (32 x 288 bf16) into LDS
#ifdef HAVE_TDM
    if (w == 0) {
        int nv0 = cnt - t0;  nv0 = nv0 > 16 ? 16 : nv0;
        tdm_gather_rows(xt, lds_offset(h_xs), toks_g + e * B, t0, nv0);
        int nv1 = cnt - t0 - 16;
        if (nv1 > 0) {
            nv1 = nv1 > 16 ? 16 : nv1;
            tdm_gather_rows(xt, lds_offset(h_xs) + 16 * KPAD * 2,
                            toks_g + e * B, t0 + 16, nv1);
        }
        __builtin_amdgcn_s_wait_tensorcnt(0);
    }
#else
    // fallback: cooperative copy, 16B chunks (36 chunks of 8 bf16 per row)
    for (int c = tid; c < TM * (KPAD / 8); c += 256) {
        int row = c / (KPAD / 8), off = (c % (KPAD / 8)) * 8;
        *reinterpret_cast<v4u*>(h_xs + row * KPAD + off) =
            *reinterpret_cast<const v4u*>(xt + (long)rows[row] * KPAD + off);
    }
#endif
    __syncthreads();

    // ---- GEMM1: staged xt (32x288) @ W1[e] -> silu -> h1s
#pragma unroll 1
    for (int nh = 0; nh < 2; ++nh) {
        int nb = w * 8 + nh * 4;
        v8f acc[2][4];
#pragma unroll
        for (int i = 0; i < 4; ++i) {
            float bv = b1[e * H + (nb + i) * 16 + col];
#pragma unroll
            for (int j = 0; j < 8; ++j) { acc[0][i][j] = bv; acc[1][i][j] = bv; }
        }
        v16bf a0c = load_a_lds(h_xs, KPAD, 0,  0, lane);
        v16bf a1c = load_a_lds(h_xs, KPAD, 16, 0, lane);
        v16bf bc[4];
#pragma unroll
        for (int i = 0; i < 4; ++i) bc[i] = load_b(W1, ((long)e * KT1 + 0) * NT_H + nb + i, lane);
#pragma unroll 2
        for (int kt = 0; kt < KT1; ++kt) {
            int ktn = (kt + 1 < KT1) ? kt + 1 : kt;
            v16bf a0n = load_a_lds(h_xs, KPAD, 0,  ktn, lane);
            v16bf a1n = load_a_lds(h_xs, KPAD, 16, ktn, lane);
            v16bf bn[4];
#pragma unroll
            for (int i = 0; i < 4; ++i)
                bn[i] = load_b(W1, ((long)e * KT1 + ktn) * NT_H + nb + i, lane);
            SCHED_BARRIER();
#pragma unroll
            for (int i = 0; i < 4; ++i) {
                acc[0][i] = wmma_bf16(a0c, bc[i], acc[0][i]);
                acc[1][i] = wmma_bf16(a1c, bc[i], acc[1][i]);
            }
            SCHED_BARRIER();
            a0c = a0n; a1c = a1n;
#pragma unroll
            for (int i = 0; i < 4; ++i) bc[i] = bn[i];
        }
#pragma unroll
        for (int mt = 0; mt < 2; ++mt)
#pragma unroll
            for (int i = 0; i < 4; ++i) {
                int n = (nb + i) * 16 + col;
#pragma unroll
                for (int v = 0; v < 8; ++v)
                    h1s[(mt * 16 + rbase + v) * H + n] = f2bf(silu_f(acc[mt][i][v]));
            }
    }
    __syncthreads();

    // ---- GEMM2: h1s (32x1024) @ W2[e] -> silu -> h2s (overwrites staged xt)
#pragma unroll 1
    for (int nh = 0; nh < 2; ++nh) {
        int nb = w * 8 + nh * 4;
        v8f acc[2][4];
#pragma unroll
        for (int i = 0; i < 4; ++i) {
            float bv = b2[e * H + (nb + i) * 16 + col];
#pragma unroll
            for (int j = 0; j < 8; ++j) { acc[0][i][j] = bv; acc[1][i][j] = bv; }
        }
        v16bf a0c = load_a_lds(h1s, H, 0,  0, lane);
        v16bf a1c = load_a_lds(h1s, H, 16, 0, lane);
        v16bf bc[4];
#pragma unroll
        for (int i = 0; i < 4; ++i) bc[i] = load_b(W2, ((long)e * KT2 + 0) * NT_H + nb + i, lane);
#pragma unroll 2
        for (int kt = 0; kt < KT2; ++kt) {
            int ktn = (kt + 1 < KT2) ? kt + 1 : kt;
            v16bf a0n = load_a_lds(h1s, H, 0,  ktn, lane);
            v16bf a1n = load_a_lds(h1s, H, 16, ktn, lane);
            v16bf bn[4];
#pragma unroll
            for (int i = 0; i < 4; ++i)
                bn[i] = load_b(W2, ((long)e * KT2 + ktn) * NT_H + nb + i, lane);
            SCHED_BARRIER();
#pragma unroll
            for (int i = 0; i < 4; ++i) {
                acc[0][i] = wmma_bf16(a0c, bc[i], acc[0][i]);
                acc[1][i] = wmma_bf16(a1c, bc[i], acc[1][i]);
            }
            SCHED_BARRIER();
            a0c = a0n; a1c = a1n;
#pragma unroll
            for (int i = 0; i < 4; ++i) bc[i] = bn[i];
        }
        __syncthreads();   // h2s aliases h_xs: ensure all reads of h1s done is NOT the issue;
                           // barrier here orders GEMM1-staging reuse across waves per nh pass
#pragma unroll
        for (int mt = 0; mt < 2; ++mt)
#pragma unroll
            for (int i = 0; i < 4; ++i) {
                int n = (nb + i) * 16 + col;
#pragma unroll
                for (int v = 0; v < 8; ++v)
                    h2s[(mt * 16 + rbase + v) * H + n] = f2bf(silu_f(acc[mt][i][v]));
            }
    }
    __syncthreads();

    // ---- GEMM3: h2s (32x1024) @ W3[e] + b3 -> alpha-weighted atomics
    {
        v8f acc[2][2];
#pragma unroll
        for (int i = 0; i < 2; ++i) {
            float bv = b3[e * D + (w * 2 + i) * 16 + col];
#pragma unroll
            for (int j = 0; j < 8; ++j) { acc[0][i][j] = bv; acc[1][i][j] = bv; }
        }
        v16bf a0c = load_a_lds(h2s, H, 0,  0, lane);
        v16bf a1c = load_a_lds(h2s, H, 16, 0, lane);
        v16bf bc[2];
#pragma unroll
        for (int i = 0; i < 2; ++i) bc[i] = load_b(W3, ((long)e * KT2 + 0) * NT_D + w * 2 + i, lane);
#pragma unroll 2
        for (int kt = 0; kt < KT2; ++kt) {
            int ktn = (kt + 1 < KT2) ? kt + 1 : kt;
            v16bf a0n = load_a_lds(h2s, H, 0,  ktn, lane);
            v16bf a1n = load_a_lds(h2s, H, 16, ktn, lane);
            v16bf bn[2];
#pragma unroll
            for (int i = 0; i < 2; ++i)
                bn[i] = load_b(W3, ((long)e * KT2 + ktn) * NT_D + w * 2 + i, lane);
            SCHED_BARRIER();
#pragma unroll
            for (int i = 0; i < 2; ++i) {
                acc[0][i] = wmma_bf16(a0c, bc[i], acc[0][i]);
                acc[1][i] = wmma_bf16(a1c, bc[i], acc[1][i]);
            }
            SCHED_BARRIER();
            a0c = a0n; a1c = a1n;
            bc[0] = bn[0]; bc[1] = bn[1];
        }
#pragma unroll
        for (int mt = 0; mt < 2; ++mt)
#pragma unroll
            for (int i = 0; i < 2; ++i) {
                int n = (w * 2 + i) * 16 + col;
#pragma unroll
                for (int v = 0; v < 8; ++v) {
                    int r = mt * 16 + rbase + v;
                    if (t0 + r < cnt)
                        atomicAdd(&out[(long)rows[r] * D + n], alph[r] * acc[mt][i][v]);
                }
            }
    }
}

// ---------------------------------------------------------------------------
// Host-side orchestration
// ---------------------------------------------------------------------------
static constexpr size_t O_XT    = 0;
static constexpr size_t O_RW1P  = O_XT    + (size_t)B * KPAD * 2;
static constexpr size_t O_EW1P  = O_RW1P  + (size_t)KPAD * H * 2;
static constexpr size_t O_EW2P  = O_EW1P  + (size_t)E * KPAD * H * 2;
static constexpr size_t O_EW3P  = O_EW2P  + (size_t)E * H * H * 2;
static constexpr size_t O_H     = O_EW3P  + (size_t)E * H * D * 2;
static constexpr size_t O_LOG   = O_H     + (size_t)B * H * 2;
static constexpr size_t O_TOK   = O_LOG   + (size_t)B * E * 4;
static constexpr size_t O_ALPHA = O_TOK   + (size_t)E * B * 4;
static constexpr size_t O_CNT   = O_ALPHA + (size_t)E * B * 4;

extern "C" void kernel_launch(void* const* d_in, const int* in_sizes, int n_in,
                              void* d_out, int out_size, void* d_ws, size_t ws_size,
                              hipStream_t stream) {
    const float* t   = (const float*)d_in[0];
    const float* x   = (const float*)d_in[1];
    const float* Rw1 = (const float*)d_in[2];
    const float* Rb1 = (const float*)d_in[3];
    const float* Rw2 = (const float*)d_in[4];
    const float* Rb2 = (const float*)d_in[5];
    const float* Eb1 = (const float*)d_in[7];
    const float* Ew1 = (const float*)d_in[6];
    const float* Ew2 = (const float*)d_in[8];
    const float* Eb2 = (const float*)d_in[9];
    const float* Ew3 = (const float*)d_in[10];
    const float* Eb3 = (const float*)d_in[11];
    float* out = (float*)d_out;

    char* ws = (char*)d_ws;
    unsigned short* xt     = (unsigned short*)(ws + O_XT);
    unsigned int*   Rw1p   = (unsigned int*)  (ws + O_RW1P);
    unsigned int*   Ew1p   = (unsigned int*)  (ws + O_EW1P);
    unsigned int*   Ew2p   = (unsigned int*)  (ws + O_EW2P);
    unsigned int*   Ew3p   = (unsigned int*)  (ws + O_EW3P);
    unsigned short* hbuf   = (unsigned short*)(ws + O_H);
    float*          logits = (float*)         (ws + O_LOG);
    int*            toks   = (int*)           (ws + O_TOK);
    float*          alphas = (float*)         (ws + O_ALPHA);
    int*            counts = (int*)           (ws + O_CNT);

    {
        long n = (long)B * D;
        k_zero<<<(int)((n + 255) / 256), 256, 0, stream>>>(out, counts, n);
    }
    {
        long n = (long)B * KPAD;
        k_pad_xt<<<(int)((n + 255) / 256), 256, 0, stream>>>(x, t, xt, n);
    }
    {
        long n = (long)1 * KT1 * NT_H * 256;
        k_repack<<<(int)((n + 255) / 256), 256, 0, stream>>>(Rw1, Rw1p, D + 1, KT1, NT_H, n);
        n = (long)E * KT1 * NT_H * 256;
        k_repack<<<(int)((n + 255) / 256), 256, 0, stream>>>(Ew1, Ew1p, D + 1, KT1, NT_H, n);
        n = (long)E * KT2 * NT_H * 256;
        k_repack<<<(int)((n + 255) / 256), 256, 0, stream>>>(Ew2, Ew2p, H, KT2, NT_H, n);
        n = (long)E * KT2 * NT_D * 256;
        k_repack<<<(int)((n + 255) / 256), 256, 0, stream>>>(Ew3, Ew3p, H, KT2, NT_D, n);
    }
    k_router_gemm<<<B / 16, 256, 0, stream>>>(xt, Rw1p, Rb1, hbuf);
    k_logits<<<B, 256, 0, stream>>>(hbuf, Rw2, Rb2, logits);
    k_topk<<<B / 256, 256, 0, stream>>>(logits, toks, alphas, counts);
    {
        size_t smem = (size_t)2 * TM * H * 2 + TM * 4 + TM * 4;
        k_expert<<<E * (B / TM), 256, smem, stream>>>(xt, Ew1p, Eb1, Ew2p, Eb2,
                                                      Ew3p, Eb3, toks, alphas,
                                                      counts, out);
    }
}
